// SCVAE_13889924235588
// MI455X (gfx1250) — matrix operand order repaired and checked by reading, not compiled, not run on hardware
//
#include <hip/hip_runtime.h>
#include <hip/hip_bf16.h>

typedef __bf16 bf16_t;
typedef __attribute__((ext_vector_type(16))) __bf16 v16bf;
typedef __attribute__((ext_vector_type(8)))  __bf16 v8bf;
typedef __attribute__((ext_vector_type(8)))  float  v8f;

#define T_STEPS 96
#define BATCH   128
#define HID     128

// activation codes
#define ACT_RELU 0
#define ACT_LEAKY 1
#define ACT_SOFTPLUS 2
#define ACT_NONE 3

__device__ __forceinline__ float act_apply(float v, int act) {
    if (act == ACT_RELU)     return fmaxf(v, 0.f);
    if (act == ACT_LEAKY)    return v > 0.f ? v : 0.1f * v;
    if (act == ACT_SOFTPLUS) return v > 20.f ? v : log1pf(expf(v));
    return v;
}
__device__ __forceinline__ float sigmoidf_(float x) { return 1.f / (1.f + expf(-x)); }

// ---------------------------------------------------------------------------
// WMMA tile machinery (16x16 output tiles, K chunks of 32, bf16 x bf16 -> f32)
// A layout per ISA 7.12.2 (16-bit A 16x32): lanes 0-15 row M=lane, K 0-7 & 16-23;
// lanes 16-31 row M=lane-16, K 8-15 & 24-31.
// B layout: lanes 0-15 col N=lane, K 0-7 & 16-23; lanes 16-31 K 8-15 & 24-31 (same k order as A)
// (== W[out][in] row-major, so the bf16 prepack is a pure dtype conversion).
// C/D: lanes 0-15 rows 0-7 (vgpr r -> M=r), lanes 16-31 rows 8-15.
// ---------------------------------------------------------------------------
struct Seg { const void* p; int lda; int w; int is_f32; };

__device__ __forceinline__ v16bf load_a_frag(const Seg& s, int row, int kh, int k0) {
    v16bf a;
    if (s.is_f32) {
        const float* r = (const float*)s.p + (size_t)row * s.lda + k0 + kh * 8;
#pragma unroll
        for (int i = 0; i < 8; ++i) { a[i] = (__bf16)r[i]; a[8 + i] = (__bf16)r[16 + i]; }
    } else {
        const bf16_t* r = (const bf16_t*)s.p + (size_t)row * s.lda + k0 + kh * 8;
        v8bf lo = *(const v8bf*)r;
        v8bf hi = *(const v8bf*)(r + 16);
#pragma unroll
        for (int i = 0; i < 8; ++i) { a[i] = lo[i]; a[8 + i] = hi[i]; }
    }
    return a;
}

// C tile = sum over concatenated segments of A[128 x K] * W[N x K]^T
__device__ __forceinline__ v8f gemm_acc(v8f acc, const Seg* segs, int nseg,
                                        const bf16_t* W, int K, int mt, int nt, int lane) {
    const int l15 = lane & 15, kh = lane >> 4;
    int kbase = 0;
#pragma unroll
    for (int s = 0; s < nseg; ++s) {
#pragma unroll
        for (int kl = 0; kl < segs[s].w; kl += 32) {
            v16bf a = load_a_frag(segs[s], mt * 16 + l15, kh, kl);
            const bf16_t* wr = W + (size_t)(nt * 16 + l15) * K + kbase + kl + kh * 8;
            v16bf b = __builtin_shufflevector(*(const v8bf*)wr, *(const v8bf*)(wr + 16), 0, 1, 2, 3, 4, 5, 6, 7, 8, 9, 10, 11, 12, 13, 14, 15);
            acc = __builtin_amdgcn_wmma_f32_16x16x32_bf16(false, a, false, b,
                                                          (short)0, acc, false, false);
        }
        kbase += segs[s].w;
    }
    return acc;
}

__device__ __forceinline__ void store_tile_bf16(bf16_t* D, int ldd, int mt, int nt, int lane,
                                                v8f c, const float* bias, int act) {
    int n = lane & 15, half = lane >> 4;
    float b = bias[nt * 16 + n];
#pragma unroll
    for (int r = 0; r < 8; ++r)
        D[(size_t)(mt * 16 + half * 8 + r) * ldd + nt * 16 + n] = (bf16_t)act_apply(c[r] + b, act);
}

__device__ __forceinline__ void store_tile_f32(float* D, int ldd, int mt, int nt, int lane,
                                               v8f c, const float* bias, int act) {
    int n = lane & 15, half = lane >> 4;
    float b = bias[nt * 16 + n];
#pragma unroll
    for (int r = 0; r < 8; ++r)
        D[(size_t)(mt * 16 + half * 8 + r) * ldd + nt * 16 + n] = act_apply(c[r] + b, act);
}

// One full [128 x N] = act([segments 128 x K] @ W^T + bias), 16 waves share tiles
__device__ __forceinline__ void phase16(const Seg* segs, int nseg, const bf16_t* W,
                                        const float* bias, int K, int N, bf16_t* dst,
                                        int act, int wave, int lane) {
    const int NT = N >> 4;
#pragma unroll
    for (int tile = wave; tile < 8 * NT; tile += 16) {
        int mt = tile / NT, nt = tile - mt * NT;
        v8f acc = {0.f, 0.f, 0.f, 0.f, 0.f, 0.f, 0.f, 0.f};
        acc = gemm_acc(acc, segs, nseg, W, K, mt, nt, lane);
        store_tile_bf16(dst, N, mt, nt, lane, acc, bias, act);
    }
}

// mean/std heads (N=16) + reparameterized sample z = leaky(m) + softplus(s)*eps
__device__ __forceinline__ void zphase(const bf16_t* feat, const bf16_t* Wm, const float* bm,
                                       const bf16_t* Ws, const float* bs, const float* eps,
                                       bf16_t* zdst /*[128][32], cols16-31 zero*/,
                                       int wave, int lane) {
    if (wave < 8) {
        Seg s[1] = {{feat, 128, 128, 0}};
        v8f am = {0.f, 0.f, 0.f, 0.f, 0.f, 0.f, 0.f, 0.f};
        v8f as2 = {0.f, 0.f, 0.f, 0.f, 0.f, 0.f, 0.f, 0.f};
        am = gemm_acc(am, s, 1, Wm, 128, wave, 0, lane);
        as2 = gemm_acc(as2, s, 1, Ws, 128, wave, 0, lane);
        int n = lane & 15, half = lane >> 4;
        float bmv = bm[n], bsv = bs[n];
#pragma unroll
        for (int r = 0; r < 8; ++r) {
            int row = wave * 16 + half * 8 + r;
            float mean = act_apply(am[r] + bmv, ACT_LEAKY);
            float sd = act_apply(as2[r] + bsv, ACT_SOFTPLUS);
            zdst[row * 32 + n] = (bf16_t)(mean + sd * eps[row * 16 + n]);
        }
    }
}

// Fused torch-GRUCell: h = (1-z)*tanh(in + r*hn) + z*h  (all 6 gate GEMMs in regs)
__device__ __forceinline__ void gru_phase(const bf16_t* phiX, const bf16_t* pz, float* H,
                                          const bf16_t* Wih, const bf16_t* Whh,
                                          const float* bih, const float* bhh,
                                          int wave, int lane) {
    float hnew[4][8];
    int n = lane & 15, half = lane >> 4;
#pragma unroll
    for (int i = 0; i < 4; ++i) {
        int tile = wave + i * 16;          // 64 tiles, 4 per wave
        int mt = tile >> 3, nt = tile & 7;
        v8f ir = {0.f,0.f,0.f,0.f,0.f,0.f,0.f,0.f}, iz = ir, in2 = ir, hr = ir, hz = ir, hn = ir;
        Seg sx[2] = {{phiX, 128, 128, 0}, {pz, 128, 128, 0}};
        Seg sh[1] = {{H, 128, 128, 1}};
        ir  = gemm_acc(ir,  sx, 2, Wih, 256, mt, nt,      lane);
        iz  = gemm_acc(iz,  sx, 2, Wih, 256, mt, nt + 8,  lane);
        in2 = gemm_acc(in2, sx, 2, Wih, 256, mt, nt + 16, lane);
        hr  = gemm_acc(hr,  sh, 1, Whh, 128, mt, nt,      lane);
        hz  = gemm_acc(hz,  sh, 1, Whh, 128, mt, nt + 8,  lane);
        hn  = gemm_acc(hn,  sh, 1, Whh, 128, mt, nt + 16, lane);
        float b_ir = bih[nt * 16 + n], b_iz = bih[128 + nt * 16 + n], b_in = bih[256 + nt * 16 + n];
        float b_hr = bhh[nt * 16 + n], b_hz = bhh[128 + nt * 16 + n], b_hn = bhh[256 + nt * 16 + n];
#pragma unroll
        for (int r = 0; r < 8; ++r) {
            int row = mt * 16 + half * 8 + r, col = nt * 16 + n;
            float rg = sigmoidf_(ir[r] + b_ir + hr[r] + b_hr);
            float zg = sigmoidf_(iz[r] + b_iz + hz[r] + b_hz);
            float nn = tanhf(in2[r] + b_in + rg * (hn[r] + b_hn));
            hnew[i][r] = (1.f - zg) * nn + zg * H[row * 128 + col];
        }
    }
    __syncthreads();   // all reads of old H done before anyone writes
#pragma unroll
    for (int i = 0; i < 4; ++i) {
        int tile = wave + i * 16;
        int mt = tile >> 3, nt = tile & 7;
#pragma unroll
        for (int r = 0; r < 8; ++r)
            H[(mt * 16 + half * 8 + r) * 128 + nt * 16 + n] = hnew[i][r];
    }
}

// ---------------------------------------------------------------------------
// Kernel 1: weight fp32 -> bf16 (with optional K zero-padding, for phi_z 16->32)
// ---------------------------------------------------------------------------
__global__ void cvt_kernel(const float* __restrict__ src, bf16_t* __restrict__ dst,
                           int rows, int inw, int outw) {
    int i = blockIdx.x * blockDim.x + threadIdx.x;
    int n = rows * outw;
    if (i >= n) return;
    int r = i / outw, c = i - r * outw;
    dst[i] = (c < inw) ? (bf16_t)src[r * inw + c] : (bf16_t)0.f;
}

// Kernel 2: embedding  out[i] = bf16(relu(dot(X[i, 0:24], w) + b))
__global__ void emb_kernel(const float* __restrict__ X, const float* __restrict__ w,
                           const float* __restrict__ b, bf16_t* __restrict__ out, int n) {
    int i = blockIdx.x * blockDim.x + threadIdx.x;
    if (i >= n) return;
    const float* r = X + (size_t)i * 24;
    float acc = b[0];
#pragma unroll
    for (int k = 0; k < 24; ++k) acc += r[k] * w[k];
    out[i] = (bf16_t)fmaxf(acc, 0.f);
}

// ---------------------------------------------------------------------------
// Kernel 3: persistent recurrent core (1 workgroup, 16 waves, loops t=0..95)
// ---------------------------------------------------------------------------
struct SeqArgs {
    const bf16_t *Xe, *Ye;
    const float *eps_enc, *eps_prior, *eps_pred;
    const bf16_t *phi_x0, *phi_x1, *phi_y0, *phi_y1, *phi_z;
    const bf16_t *enc0, *enc1, *enc_mean, *enc_std;
    const bf16_t *prior0, *prior1, *prior_mean, *prior_std;
    const bf16_t *pred0, *pred1, *pred_mean, *pred_std;
    const bf16_t *dec0, *dec1, *decp0, *decp1, *decq0, *decq1;
    const bf16_t *rnn_Wih, *rnn_Whh, *rnn2_Wih, *rnn2_Whh;
    const float *b_phi_x0, *b_phi_x1, *b_phi_y0, *b_phi_y1, *b_phi_z;
    const float *b_enc0, *b_enc1, *b_enc_mean, *b_enc_std;
    const float *b_prior0, *b_prior1, *b_prior_mean, *b_prior_std;
    const float *b_pred0, *b_pred1, *b_pred_mean, *b_pred_std;
    const float *b_dec0, *b_dec1, *b_decp0, *b_decp1, *b_decq0, *b_decq1;
    const float *rnn_bih, *rnn_bhh, *rnn2_bih, *rnn2_bhh;
    bf16_t *featDec, *featDecp, *featDecq;
};

__global__ __launch_bounds__(512, 1) void seq_kernel(SeqArgs a) {
    __shared__ __align__(16) bf16_t sPhiX[128 * 128];
    __shared__ __align__(16) bf16_t sT0[128 * 128];
    __shared__ __align__(16) bf16_t sT1[128 * 128];
    __shared__ __align__(16) bf16_t sPz[128 * 128];
    __shared__ __align__(16) bf16_t sZ0[128 * 32], sZ1[128 * 32], sZ2[128 * 32];
    __shared__ __align__(16) float sH[128 * 128], sH2[128 * 128];

    const int tid = threadIdx.x, lane = tid & 31;
    const int wave = __builtin_amdgcn_readfirstlane(tid >> 5);  // scalar -> scalar loop control
    for (int i = tid; i < 128 * 128; i += 512) { sH[i] = 0.f; sH2[i] = 0.f; }
    for (int i = tid; i < 128 * 32; i += 512) {
        sZ0[i] = (bf16_t)0.f; sZ1[i] = (bf16_t)0.f; sZ2[i] = (bf16_t)0.f;
    }
    __syncthreads();

    for (int t = 0; t < T_STEPS; ++t) {
        // prefetch next step's only cold global inputs (weights are L2-resident)
        if (t + 1 < T_STEPS) {
            const char* px = (const char*)(a.Xe + (size_t)(t + 1) * BATCH * 64);
            const char* py = (const char*)(a.Ye + (size_t)(t + 1) * BATCH * 32);
            if (tid < 128) __builtin_prefetch(px + tid * 128, 0, 0);
            if (tid >= 128 && tid < 192) __builtin_prefetch(py + (tid - 128) * 128, 0, 0);
            const char* pe = (const char*)(a.eps_enc + (size_t)(t + 1) * BATCH * 16);
            const char* pp = (const char*)(a.eps_prior + (size_t)(t + 1) * BATCH * 16);
            const char* pq = (const char*)(a.eps_pred + (size_t)(t + 1) * BATCH * 16);
            if (tid >= 192 && tid < 256) __builtin_prefetch(pe + (tid - 192) * 128, 0, 0);
            if (tid >= 256 && tid < 320) __builtin_prefetch(pp + (tid - 256) * 128, 0, 0);
            if (tid >= 320 && tid < 384) __builtin_prefetch(pq + (tid - 320) * 128, 0, 0);
        }
        const float* epsE = a.eps_enc   + (size_t)t * BATCH * 16;
        const float* epsP = a.eps_prior + (size_t)t * BATCH * 16;
        const float* epsQ = a.eps_pred  + (size_t)t * BATCH * 16;

        // phi_x = mlp2(x_emb)
        { Seg s[1] = {{a.Xe + (size_t)t * BATCH * 64, 64, 64, 0}};
          phase16(s, 1, a.phi_x0, a.b_phi_x0, 64, 128, sT0, ACT_RELU, wave, lane); } __syncthreads();
        { Seg s[1] = {{sT0, 128, 128, 0}};
          phase16(s, 1, a.phi_x1, a.b_phi_x1, 128, 128, sPhiX, ACT_RELU, wave, lane); } __syncthreads();
        // phi_y = mlp2(y_emb)  -> sT1
        { Seg s[1] = {{a.Ye + (size_t)t * BATCH * 32, 32, 32, 0}};
          phase16(s, 1, a.phi_y0, a.b_phi_y0, 32, 128, sT0, ACT_RELU, wave, lane); } __syncthreads();
        { Seg s[1] = {{sT0, 128, 128, 0}};
          phase16(s, 1, a.phi_y1, a.b_phi_y1, 128, 128, sT1, ACT_RELU, wave, lane); } __syncthreads();
        // enc = mlp2([phi_x, phi_y, h])
        { Seg s[3] = {{sPhiX, 128, 128, 0}, {sT1, 128, 128, 0}, {sH, 128, 128, 1}};
          phase16(s, 3, a.enc0, a.b_enc0, 384, 128, sT0, ACT_RELU, wave, lane); } __syncthreads();
        { Seg s[1] = {{sT0, 128, 128, 0}};
          phase16(s, 1, a.enc1, a.b_enc1, 128, 128, sT1, ACT_RELU, wave, lane); } __syncthreads();
        zphase(sT1, a.enc_mean, a.b_enc_mean, a.enc_std, a.b_enc_std, epsE, sZ0, wave, lane); __syncthreads();
        // prior = mlp2([phi_x, h])
        { Seg s[2] = {{sPhiX, 128, 128, 0}, {sH, 128, 128, 1}};
          phase16(s, 2, a.prior0, a.b_prior0, 256, 128, sT0, ACT_RELU, wave, lane); } __syncthreads();
        { Seg s[1] = {{sT0, 128, 128, 0}};
          phase16(s, 1, a.prior1, a.b_prior1, 128, 128, sT1, ACT_RELU, wave, lane); } __syncthreads();
        zphase(sT1, a.prior_mean, a.b_prior_mean, a.prior_std, a.b_prior_std, epsP, sZ1, wave, lane); __syncthreads();
        // pred = mlp2([phi_x, h2])
        { Seg s[2] = {{sPhiX, 128, 128, 0}, {sH2, 128, 128, 1}};
          phase16(s, 2, a.pred0, a.b_pred0, 256, 128, sT0, ACT_RELU, wave, lane); } __syncthreads();
        { Seg s[1] = {{sT0, 128, 128, 0}};
          phase16(s, 1, a.pred1, a.b_pred1, 128, 128, sT1, ACT_RELU, wave, lane); } __syncthreads();
        zphase(sT1, a.pred_mean, a.b_pred_mean, a.pred_std, a.b_pred_std, epsQ, sZ2, wave, lane); __syncthreads();
        // pz (posterior) = relu(lin(z, phi_z)), K padded 16->32
        { Seg s[1] = {{sZ0, 32, 32, 0}};
          phase16(s, 1, a.phi_z, a.b_phi_z, 32, 128, sPz, ACT_RELU, wave, lane); } __syncthreads();
        // dec feat = mlp2([phi_x, pz, h]) -> ws
        { Seg s[3] = {{sPhiX, 128, 128, 0}, {sPz, 128, 128, 0}, {sH, 128, 128, 1}};
          phase16(s, 3, a.dec0, a.b_dec0, 384, 128, sT0, ACT_RELU, wave, lane); } __syncthreads();
        { Seg s[1] = {{sT0, 128, 128, 0}};
          phase16(s, 1, a.dec1, a.b_dec1, 128, 128, a.featDec + (size_t)t * BATCH * HID,
                  ACT_RELU, wave, lane); } __syncthreads();
        // pz_prior -> sT0 ; decp feat = mlp2([phi_x, pz_pr, h]) -> ws
        { Seg s[1] = {{sZ1, 32, 32, 0}};
          phase16(s, 1, a.phi_z, a.b_phi_z, 32, 128, sT0, ACT_RELU, wave, lane); } __syncthreads();
        { Seg s[3] = {{sPhiX, 128, 128, 0}, {sT0, 128, 128, 0}, {sH, 128, 128, 1}};
          phase16(s, 3, a.decp0, a.b_decp0, 384, 128, sT1, ACT_RELU, wave, lane); } __syncthreads();
        { Seg s[1] = {{sT1, 128, 128, 0}};
          phase16(s, 1, a.decp1, a.b_decp1, 128, 128, a.featDecp + (size_t)t * BATCH * HID,
                  ACT_RELU, wave, lane); } __syncthreads();
        // GRU1: h = GRU([phi_x, pz], h)   (last consumer of old h)
        gru_phase(sPhiX, sPz, sH, a.rnn_Wih, a.rnn_Whh, a.rnn_bih, a.rnn_bhh, wave, lane); __syncthreads();
        // pz_pred -> sT0 ; decq feat = mlp2([phi_x, pz_pd, h2]) -> ws
        { Seg s[1] = {{sZ2, 32, 32, 0}};
          phase16(s, 1, a.phi_z, a.b_phi_z, 32, 128, sT0, ACT_RELU, wave, lane); } __syncthreads();
        { Seg s[3] = {{sPhiX, 128, 128, 0}, {sT0, 128, 128, 0}, {sH2, 128, 128, 1}};
          phase16(s, 3, a.decq0, a.b_decq0, 384, 128, sT1, ACT_RELU, wave, lane); } __syncthreads();
        { Seg s[1] = {{sT1, 128, 128, 0}};
          phase16(s, 1, a.decq1, a.b_decq1, 128, 128, a.featDecq + (size_t)t * BATCH * HID,
                  ACT_RELU, wave, lane); } __syncthreads();
        // GRU2: h2 = GRU([phi_x, pz_pd], h2)
        gru_phase(sPhiX, sT0, sH2, a.rnn2_Wih, a.rnn2_Whh, a.rnn2_bih, a.rnn2_bhh, wave, lane); __syncthreads();
    }
}

// ---------------------------------------------------------------------------
// Kernel 4: decoder heads, fully parallel: 3 paths x 96 t x 8 row-tiles
// dm = leaky(lin(leaky(lin(feat, m0)), m1)) ; ds = softplus(lin(feat, std))
// ---------------------------------------------------------------------------
__global__ __launch_bounds__(256, 2) void heads_kernel(
    const bf16_t* __restrict__ feats, const bf16_t* __restrict__ Wm0, const float* __restrict__ bm0,
    const bf16_t* __restrict__ Wm1, const float* __restrict__ bm1,
    const bf16_t* __restrict__ Wsd, const float* __restrict__ bsd, float* __restrict__ out) {
    __shared__ __align__(16) bf16_t sM0[16 * 128];
    int lane = threadIdx.x & 31;
    int wave = __builtin_amdgcn_readfirstlane(threadIdx.x >> 5);   // 8 waves
    int blk = blockIdx.x;
    int p = blk / (T_STEPS * 8);
    int rem = blk - p * (T_STEPS * 8);
    int t = rem >> 3, mt = rem & 7;
    const bf16_t* feat = feats + ((size_t)(p * T_STEPS + t)) * BATCH * HID;

    // stage 1: m0[16 x 128] = leaky(feat_tile @ Wm0^T + b)
    { Seg s[1] = {{feat, 128, 128, 0}};
      v8f acc = {0.f, 0.f, 0.f, 0.f, 0.f, 0.f, 0.f, 0.f};
      acc = gemm_acc(acc, s, 1, Wm0, 128, mt, wave, lane);
      store_tile_bf16(sM0, 128, 0, wave, lane, acc, bm0, ACT_LEAKY); }
    __syncthreads();
    // stage 2: dm[16 x 768]
    float* dm = out + ((size_t)((2 * p) * T_STEPS + t)) * BATCH * 768;
#pragma unroll
    for (int nt = wave; nt < 48; nt += 8) {
        Seg s[1] = {{sM0, 128, 128, 0}};
        v8f acc = {0.f, 0.f, 0.f, 0.f, 0.f, 0.f, 0.f, 0.f};
        acc = gemm_acc(acc, s, 1, Wm1, 128, 0, nt, lane);
        store_tile_f32(dm, 768, mt, nt, lane, acc, bm1, ACT_LEAKY);
    }
    // stage 3: ds[16 x 768]
    float* dsd = out + ((size_t)((2 * p + 1) * T_STEPS + t)) * BATCH * 768;
#pragma unroll
    for (int nt = wave; nt < 48; nt += 8) {
        Seg s[1] = {{feat, 128, 128, 0}};
        v8f acc = {0.f, 0.f, 0.f, 0.f, 0.f, 0.f, 0.f, 0.f};
        acc = gemm_acc(acc, s, 1, Wsd, 128, mt, nt, lane);
        store_tile_f32(dsd, 768, mt, nt, lane, acc, bsd, ACT_SOFTPLUS);
    }
}

// ---------------------------------------------------------------------------
// Host launch
// ---------------------------------------------------------------------------
struct CvtSpec { int idx, rows, inw, outw; };

extern "C" void kernel_launch(void* const* d_in, const int* in_sizes, int n_in,
                              void* d_out, int out_size, void* d_ws, size_t ws_size,
                              hipStream_t stream) {
    (void)in_sizes; (void)n_in; (void)out_size; (void)ws_size;
    char* ws = (char*)d_ws;
    size_t off = 0;
    auto alloc = [&](size_t bytes) -> void* {
        void* r = ws + off;
        off += (bytes + 255) & ~(size_t)255;
        return r;
    };
    bf16_t* Xe = (bf16_t*)alloc((size_t)T_STEPS * BATCH * 64 * 2);
    bf16_t* Ye = (bf16_t*)alloc((size_t)T_STEPS * BATCH * 32 * 2);
    bf16_t* featAll = (bf16_t*)alloc((size_t)3 * T_STEPS * BATCH * HID * 2);

    static const CvtSpec cv[30] = {
        {7, 128, 64, 64},   {9, 128, 128, 128}, {11, 128, 32, 32},  {13, 128, 128, 128},
        {15, 128, 16, 32},  {17, 128, 384, 384},{19, 128, 128, 128},{21, 16, 128, 128},
        {23, 16, 128, 128}, {25, 128, 256, 256},{27, 128, 128, 128},{29, 16, 128, 128},
        {31, 16, 128, 128}, {33, 128, 256, 256},{35, 128, 128, 128},{37, 16, 128, 128},
        {39, 16, 128, 128}, {41, 128, 384, 384},{43, 128, 128, 128},{45, 128, 384, 384},
        {47, 128, 128, 128},{49, 128, 384, 384},{51, 128, 128, 128},{53, 128, 128, 128},
        {55, 768, 128, 128},{57, 768, 128, 128},{59, 384, 256, 256},{60, 384, 128, 128},
        {63, 384, 256, 256},{64, 384, 128, 128}};
    bf16_t* wp[30];
    for (int i = 0; i < 30; ++i)
        wp[i] = (bf16_t*)alloc((size_t)cv[i].rows * cv[i].outw * 2);
    for (int i = 0; i < 30; ++i) {
        int n = cv[i].rows * cv[i].outw;
        cvt_kernel<<<(n + 255) / 256, 256, 0, stream>>>(
            (const float*)d_in[cv[i].idx], wp[i], cv[i].rows, cv[i].inw, cv[i].outw);
    }
    int nx = T_STEPS * BATCH * 64, ny = T_STEPS * BATCH * 32;
    emb_kernel<<<(nx + 255) / 256, 256, 0, stream>>>(
        (const float*)d_in[0], (const float*)d_in[5], (const float*)d_in[6], Xe, nx);
    emb_kernel<<<(ny + 255) / 256, 256, 0, stream>>>(
        (const float*)d_in[1], (const float*)d_in[5], (const float*)d_in[6], Ye, ny);

    SeqArgs a;
    a.Xe = Xe; a.Ye = Ye;
    a.eps_enc = (const float*)d_in[2];
    a.eps_prior = (const float*)d_in[3];
    a.eps_pred = (const float*)d_in[4];
    a.phi_x0 = wp[0];  a.phi_x1 = wp[1];  a.phi_y0 = wp[2];  a.phi_y1 = wp[3];  a.phi_z = wp[4];
    a.enc0 = wp[5];    a.enc1 = wp[6];    a.enc_mean = wp[7];  a.enc_std = wp[8];
    a.prior0 = wp[9];  a.prior1 = wp[10]; a.prior_mean = wp[11]; a.prior_std = wp[12];
    a.pred0 = wp[13];  a.pred1 = wp[14];  a.pred_mean = wp[15];  a.pred_std = wp[16];
    a.dec0 = wp[17];   a.dec1 = wp[18];   a.decp0 = wp[19];  a.decp1 = wp[20];
    a.decq0 = wp[21];  a.decq1 = wp[22];
    a.rnn_Wih = wp[26]; a.rnn_Whh = wp[27]; a.rnn2_Wih = wp[28]; a.rnn2_Whh = wp[29];
    a.b_phi_x0 = (const float*)d_in[8];  a.b_phi_x1 = (const float*)d_in[10];
    a.b_phi_y0 = (const float*)d_in[12]; a.b_phi_y1 = (const float*)d_in[14];
    a.b_phi_z = (const float*)d_in[16];
    a.b_enc0 = (const float*)d_in[18];   a.b_enc1 = (const float*)d_in[20];
    a.b_enc_mean = (const float*)d_in[22]; a.b_enc_std = (const float*)d_in[24];
    a.b_prior0 = (const float*)d_in[26]; a.b_prior1 = (const float*)d_in[28];
    a.b_prior_mean = (const float*)d_in[30]; a.b_prior_std = (const float*)d_in[32];
    a.b_pred0 = (const float*)d_in[34];  a.b_pred1 = (const float*)d_in[36];
    a.b_pred_mean = (const float*)d_in[38]; a.b_pred_std = (const float*)d_in[40];
    a.b_dec0 = (const float*)d_in[42];   a.b_dec1 = (const float*)d_in[44];
    a.b_decp0 = (const float*)d_in[46];  a.b_decp1 = (const float*)d_in[48];
    a.b_decq0 = (const float*)d_in[50];  a.b_decq1 = (const float*)d_in[52];
    a.rnn_bih = (const float*)d_in[61];  a.rnn_bhh = (const float*)d_in[62];
    a.rnn2_bih = (const float*)d_in[65]; a.rnn2_bhh = (const float*)d_in[66];
    a.featDec = featAll;
    a.featDecp = featAll + (size_t)T_STEPS * BATCH * HID;
    a.featDecq = featAll + (size_t)2 * T_STEPS * BATCH * HID;

    seq_kernel<<<1, 512, 0, stream>>>(a);

    heads_kernel<<<3 * T_STEPS * 8, 256, 0, stream>>>(
        featAll, wp[23], (const float*)d_in[54], wp[24], (const float*)d_in[56],
        wp[25], (const float*)d_in[58], (float*)d_out);
}
